// ParallelMultiHeadSelfAttention_74655121539566
// MI455X (gfx1250) — matrix operand order repaired and checked by reading, not compiled, not run on hardware
//
#include <hip/hip_runtime.h>
#include <hip/hip_bf16.h>

typedef __attribute__((ext_vector_type(16))) _Float16 v16h;
typedef __attribute__((ext_vector_type(8)))  float    v8f;

#define NB    2      // batch
#define NH    12     // heads
#define DD    64     // head dim
#define NSEQ  2048   // sequence length
#define KIN   768    // embed
#define NOUT  2304   // 3*H*D

static __device__ __forceinline__ v8f wmma16(const v16h& a, const v16h& b, const v8f& c) {
  return __builtin_amdgcn_wmma_f32_16x16x32_f16(false, a, false, b, (short)0, c, false, false);
}

// ---------------------------------------------------------------------------
// Kernel 1: fused QKV projection.  C[m][n] = z[m][:]·W[:][n] + bias[n]
// One wave per 16x64 tile (4 accumulators share one A fragment), K in chunks
// of 32 via v_wmma_f32_16x16x32_f16.  global_prefetch_b8 pulls the next
// K-chunk of z/W while the current WMMAs run.
// Scatter epilogue: column n -> (h = n/192, d = (n/3)%64, s = n%3);
//   s==0 -> K  stored [bh][key][d]   (f16)
//   s==1 -> V  stored [bh][d][key]   (f16, transposed for AV B-fragments)
//   s==2 -> Q  stored [bh][query][d] (f16)
// ---------------------------------------------------------------------------
__global__ void qkv_gemm_kernel(const float* __restrict__ z,
                                const float* __restrict__ W,
                                const float* __restrict__ bias,
                                _Float16* __restrict__ Qf,
                                _Float16* __restrict__ Kf,
                                _Float16* __restrict__ Vt) {
  const int lane = threadIdx.x & 31;
  const int wave = threadIdx.x >> 5;
  const int w    = blockIdx.x * 4 + wave;      // (NB*NSEQ/16)*(NOUT/64) waves
  const int mt   = w / (NOUT / 64);
  const int nt   = w % (NOUT / 64);
  const int row0 = mt * 16;
  const int col0 = nt * 64;
  const int lm   = lane & 15;
  const int hi   = lane >> 4;

  const int m = row0 + lm;                     // A-fragment row

  v8f acc0 = {}, acc1 = {}, acc2 = {}, acc3 = {};
  for (int kk = 0; kk < KIN; kk += 32) {
    // prefetch next K-chunk of this z row (skipped past end: harmless hint)
    if (kk + 32 < KIN) {
      __builtin_prefetch(&z[(size_t)m * KIN + kk + 32], 0, 1);
      __builtin_prefetch(&W[(size_t)(kk + 32 + (hi << 4)) * NOUT + col0 + lm], 0, 1);
    }
    v16h a;
#pragma unroll
    for (int v = 0; v < 8; ++v) {
      // A 16x32 f16 layout: k0 = (v>=4?16:0) + hi*8 + (v&3)*2
      const int k0 = kk + ((v >> 2) << 4) + (hi << 3) + ((v & 3) << 1);
      a[2 * v]     = (_Float16)z[(size_t)m * KIN + k0];
      a[2 * v + 1] = (_Float16)z[(size_t)m * KIN + k0 + 1];
    }
#pragma unroll
    for (int t = 0; t < 4; ++t) {
      const int n = col0 + t * 16 + lm;        // B column (coalesced across lanes)
      v16h b;
#pragma unroll
      for (int v = 0; v < 8; ++v) {
        // B 32x16 f16 layout: k = (v>=4?16:0) + hi*8 + (v&3)*2
        const int kb = kk + ((v >> 2) << 4) + (hi << 3) + ((v & 3) << 1);
        b[2 * v]     = (_Float16)W[(size_t)kb * NOUT + n];
        b[2 * v + 1] = (_Float16)W[(size_t)(kb + 1) * NOUT + n];
      }
      if (t == 0)      acc0 = wmma16(a, b, acc0);
      else if (t == 1) acc1 = wmma16(a, b, acc1);
      else if (t == 2) acc2 = wmma16(a, b, acc2);
      else             acc3 = wmma16(a, b, acc3);
    }
  }

#pragma unroll
  for (int t = 0; t < 4; ++t) {
    const v8f& acc = (t == 0) ? acc0 : (t == 1) ? acc1 : (t == 2) ? acc2 : acc3;
    const int n = col0 + t * 16 + lm;
    const float bv = bias[n];
    const int s = n % 3;
    const int d = (n / 3) % DD;
    const int h = n / (3 * DD);
#pragma unroll
    for (int r = 0; r < 8; ++r) {
      const int gm = row0 + r + hi * 8;        // C layout: M = r + hi*8
      const int bb = gm >> 11;                 // / NSEQ
      const int nn = gm & (NSEQ - 1);
      const int bh = bb * NH + h;
      const _Float16 hv = (_Float16)(acc[r] + bv);
      if (s == 0)       Kf[((size_t)bh * NSEQ + nn) * DD + d] = hv;
      else if (s == 1)  Vt[((size_t)bh * DD + d) * NSEQ + nn] = hv;
      else              Qf[((size_t)bh * NSEQ + nn) * DD + d] = hv;
    }
  }
}

// ---------------------------------------------------------------------------
// Kernel 2: flash-style attention per (b,h,16-query tile), one wave per block.
// Key loop in tiles of 32; online softmax with wave32 butterfly reductions
// (masks 1/2/4/8 stay inside each 16-lane C-row group).  Reproduces the
// reference bug: probabilities scaled by D^-0.5 AFTER softmax.
// ---------------------------------------------------------------------------
__global__ void attn_kernel(const _Float16* __restrict__ Qf,
                            const _Float16* __restrict__ Kf,
                            const _Float16* __restrict__ Vt,
                            float* __restrict__ out) {
  __shared__ _Float16 lds_p[16][32];   // P tile bounce: C-layout -> A-layout

  const int lane = threadIdx.x & 31;
  const int lm   = lane & 15;
  const int hi   = lane >> 4;
  const int qt   = blockIdx.x % (NSEQ / 16);
  const int bh   = blockIdx.x / (NSEQ / 16);
  const int bb   = bh / NH;
  const int h    = bh % NH;

  const _Float16* Qp = Qf + (size_t)bh * NSEQ * DD;
  const _Float16* Kp = Kf + (size_t)bh * NSEQ * DD;
  const _Float16* Vp = Vt + (size_t)bh * DD * NSEQ;

  // Q A-fragments for this 16-row tile, d split into two 32-chunks
  v16h aq[2];
  const int qrow = qt * 16 + lm;
#pragma unroll
  for (int c = 0; c < 2; ++c)
#pragma unroll
    for (int v = 0; v < 8; ++v) {
      const int k0 = c * 32 + ((v >> 2) << 4) + (hi << 3) + ((v & 3) << 1);
      aq[c][2 * v]     = Qp[(size_t)qrow * DD + k0];
      aq[c][2 * v + 1] = Qp[(size_t)qrow * DD + k0 + 1];
    }

  float mrow[8], lrow[8];
  v8f   o[4];
#pragma unroll
  for (int r = 0; r < 8; ++r) { mrow[r] = -1e30f; lrow[r] = 0.0f; }
#pragma unroll
  for (int t = 0; t < 4; ++t) o[t] = (v8f){};

  for (int j = 0; j < NSEQ / 32; ++j) {
    // prefetch next key tile's K rows and V columns while this tile computes
    if (j + 1 < NSEQ / 32) {
      __builtin_prefetch(&Kp[(size_t)((j + 1) * 32 + lane) * DD], 0, 1);
      __builtin_prefetch(&Vp[(size_t)lane * NSEQ * 2 + (j + 1) * 32], 0, 1);
    }

    // ---- scores S = Q · K^T for 32 keys (two 16-key sub-tiles) ----
    v8f s0 = {}, s1 = {};
#pragma unroll
    for (int c = 0; c < 2; ++c) {
      v16h bk0, bk1;
      const int key0 = j * 32 + lm;
      const int key1 = key0 + 16;
#pragma unroll
      for (int v = 0; v < 8; ++v) {
        const int dd = c * 32 + ((v >> 2) << 4) + (hi << 3) + ((v & 3) << 1);  // B k-dim = d, pairs contiguous
        bk0[2 * v]     = Kp[(size_t)key0 * DD + dd];
        bk0[2 * v + 1] = Kp[(size_t)key0 * DD + dd + 1];
        bk1[2 * v]     = Kp[(size_t)key1 * DD + dd];
        bk1[2 * v + 1] = Kp[(size_t)key1 * DD + dd + 1];
      }
      s0 = wmma16(aq[c], bk0, s0);
      s1 = wmma16(aq[c], bk1, s1);
    }

    // ---- online softmax statistics (per C-row r, reduced over 16 lanes) ----
    float mnew[8], scl[8], p0[8], p1[8];
#pragma unroll
    for (int r = 0; r < 8; ++r) {
      float v = fmaxf(s0[r], s1[r]);
      v = fmaxf(v, __shfl_xor(v, 1));
      v = fmaxf(v, __shfl_xor(v, 2));
      v = fmaxf(v, __shfl_xor(v, 4));
      v = fmaxf(v, __shfl_xor(v, 8));
      mnew[r] = fmaxf(mrow[r], v);
      scl[r]  = __expf(mrow[r] - mnew[r]);
      mrow[r] = mnew[r];
      p0[r] = __expf(s0[r] - mnew[r]);
      p1[r] = __expf(s1[r] - mnew[r]);
      float rs = p0[r] + p1[r];
      rs += __shfl_xor(rs, 1);
      rs += __shfl_xor(rs, 2);
      rs += __shfl_xor(rs, 4);
      rs += __shfl_xor(rs, 8);
      lrow[r] = lrow[r] * scl[r] + rs;
    }

    // ---- restripe P from C-layout to A-layout via LDS ----
    __syncthreads();   // WAR vs previous iteration's LDS reads
#pragma unroll
    for (int r = 0; r < 8; ++r) {
      const int m = r + hi * 8;
      lds_p[m][lm]      = (_Float16)p0[r];
      lds_p[m][16 + lm] = (_Float16)p1[r];
    }
    __syncthreads();
    v16h ap;
#pragma unroll
    for (int v = 0; v < 8; ++v) {
      const int k0 = ((v >> 2) << 4) + (hi << 3) + ((v & 3) << 1);
      ap[2 * v]     = lds_p[lm][k0];
      ap[2 * v + 1] = lds_p[lm][k0 + 1];
    }

    // ---- O = O*scale + P·V  (V transposed: (key,key+1) pairs contiguous) ----
#pragma unroll
    for (int t = 0; t < 4; ++t) {
#pragma unroll
      for (int r = 0; r < 8; ++r) o[t][r] *= scl[r];
      v16h bv;
      const int dd = t * 16 + lm;
#pragma unroll
      for (int v = 0; v < 8; ++v) {
        const int key = j * 32 + ((v >> 2) << 4) + (hi << 3) + ((v & 3) << 1);
        bv[2 * v]     = Vp[(size_t)dd * NSEQ + key];
        bv[2 * v + 1] = Vp[(size_t)dd * NSEQ + key + 1];
      }
      o[t] = wmma16(ap, bv, o[t]);
    }
  }

  // ---- epilogue: out[b][q][h*64+d] = O * (D^-0.5 / l)  (post-softmax scale bug) ----
#pragma unroll
  for (int r = 0; r < 8; ++r) {
    const float f  = 0.125f / lrow[r];           // 64^-0.5 * 1/l
    const int   qq = qt * 16 + r + hi * 8;
#pragma unroll
    for (int t = 0; t < 4; ++t) {
      const int dd = t * 16 + lm;
      out[((size_t)bb * NSEQ + qq) * (NH * DD) + h * DD + dd] = o[t][r] * f;
    }
  }
}

extern "C" void kernel_launch(void* const* d_in, const int* in_sizes, int n_in,
                              void* d_out, int out_size, void* d_ws, size_t ws_size,
                              hipStream_t stream) {
  const float* z    = (const float*)d_in[0];
  const float* W    = (const float*)d_in[1];
  const float* bias = (const float*)d_in[2];
  float* out = (float*)d_out;

  const size_t per_buf = (size_t)NB * NH * NSEQ * DD;   // f16 elements each
  _Float16* Qf = (_Float16*)d_ws;
  _Float16* Kf = Qf + per_buf;
  _Float16* Vt = Kf + per_buf;

  // GEMM: (NB*NSEQ/16) * (NOUT/64) = 256*36 = 9216 wave-tiles, 4 waves/block
  const int gemm_blocks = (NB * NSEQ / 16) * (NOUT / 64) / 4;   // 2304
  qkv_gemm_kernel<<<gemm_blocks, 128, 0, stream>>>(z, W, bias, Qf, Kf, Vt);

  // Attention: NB*NH*(NSEQ/16) = 3072 single-wave workgroups
  const int attn_blocks = NB * NH * (NSEQ / 16);
  attn_kernel<<<attn_blocks, 32, 0, stream>>>(Qf, Kf, Vt, out);
}
